// UGS_56994216018168
// MI455X (gfx1250) — matrix operand order repaired and checked by reading, not compiled, not run on hardware
//
#include <hip/hip_runtime.h>
#include <hip/hip_bf16.h>

#define N_NODES 100000
#define N_EDGES 1600000
#define IN_DIM 128
#define HID 128
#define OUT_DIM 64

typedef __attribute__((ext_vector_type(16))) __bf16 v16bf;
typedef __attribute__((ext_vector_type(8)))  float  v8f;

// ---------------------------------------------------------------------------
// Edge scatter-add: agg[dst] += feat[src].  One wave32 per edge.
//   DIM=128: lane owns 4 floats (float4 gather, 4 f32 atomics)
//   DIM=64 : lane owns 2 floats (float2 gather, 2 f32 atomics)
// feat/agg fit in the 192MB L2, so gathers and atomics stay on-chip.
// ---------------------------------------------------------------------------
template <int DIM>
__global__ __launch_bounds__(256) void sage_scatter_add(
    const float* __restrict__ feat,
    const long long* __restrict__ src,
    const long long* __restrict__ dst,
    float* __restrict__ agg,
    int n_edges)
{
    int gid  = blockIdx.x * blockDim.x + threadIdx.x;
    int edge = gid >> 5;
    int lane = gid & 31;
    if (edge >= n_edges) return;
    long long s = src[edge];
    long long d = dst[edge];
    if constexpr (DIM == 128) {
        const float4 v = *(const float4*)(feat + (size_t)s * 128 + lane * 4);
        float* out = agg + (size_t)d * 128 + lane * 4;
        atomicAdd(out + 0, v.x);
        atomicAdd(out + 1, v.y);
        atomicAdd(out + 2, v.z);
        atomicAdd(out + 3, v.w);
    } else {
        const float2 v = *(const float2*)(feat + (size_t)s * 64 + lane * 2);
        float* out = agg + (size_t)d * 64 + lane * 2;
        atomicAdd(out + 0, v.x);
        atomicAdd(out + 1, v.y);
    }
}

// ---------------------------------------------------------------------------
// Tiled GEMM with bf16 WMMA, f32 accumulate.
//   C = epilogue( [A0 | A1] @ [B0 ; B1] (+ Cadd) (+ bias) )
// KSRC = number of 128-wide K sources (1 or 2).  Block = 128 thr = 4 waves,
// M-tile 64 (wave w owns rows w*16..+15), K streamed 32 at a time via LDS.
//   Alds: row-major [64][36] (pad -> 18 banks/row)
//   Blds: K-pair-contiguous per column [NOUT][34] -> B frag = 16 contig bf16
// B fragments preloaded 4 tiles at a time so WMMAs issue back-to-back.
// EPI: 0 = bias+ReLU, 1 = plain store (no bias), 2 = bias+addend+softmax64.
// ---------------------------------------------------------------------------
template <int NOUT, int KSRC, int EPI>
__global__ __launch_bounds__(128) void sage_gemm(
    const float* __restrict__ A0, const float* __restrict__ A1,
    const float* __restrict__ B0, const float* __restrict__ B1,
    const float* __restrict__ bias,
    const float* __restrict__ Cadd,
    float* __restrict__ C,
    int nrows)
{
    constexpr int NT = NOUT / 16;
    static_assert(EPI != 2 || NOUT == 64, "softmax epilogue needs full row per wave");
    __shared__ __bf16 Alds[64 * 36];
    __shared__ __bf16 Blds[NOUT * 34];

    const int t     = threadIdx.x;
    const int lane  = t & 31;
    const int wv    = t >> 5;
    const int m16   = lane & 15;
    const int half  = lane >> 4;
    const int mbase = blockIdx.x * 64;

    v8f acc[NT];
#pragma unroll
    for (int nt = 0; nt < NT; ++nt) acc[nt] = (v8f)0.0f;

#pragma unroll 1
    for (int ks = 0; ks < KSRC * 4; ++ks) {
        const float* Asrc = (KSRC == 2 && ks >= 4) ? A1 : A0;
        const float* Bsrc = (KSRC == 2 && ks >= 4) ? B1 : B0;
        const int koff = (ks & 3) * 32;

        // ---- A tile: 64 rows x 32 k, f32 -> bf16 ----
#pragma unroll
        for (int idx = 0; idx < 4; ++idx) {
            int chunk = t + idx * 128;
            int r  = chunk >> 3;
            int c4 = (chunk & 7) << 2;
            int grow = mbase + r;
            float4 v = make_float4(0.f, 0.f, 0.f, 0.f);
            if (grow < nrows)
                v = *(const float4*)(Asrc + (size_t)grow * 128 + koff + c4);
            __bf16* dp = &Alds[r * 36 + c4];
            dp[0] = (__bf16)v.x; dp[1] = (__bf16)v.y;
            dp[2] = (__bf16)v.z; dp[3] = (__bf16)v.w;
        }

        // ---- B tile: 32 k x NOUT, transposed to column-pair-contiguous ----
        constexpr int BELEMS = 32 * NOUT;
#pragma unroll
        for (int idx = t; idx < BELEMS; idx += 128) {
            int kk = idx / NOUT;
            int n  = idx % NOUT;
            Blds[n * 34 + kk] = (__bf16)Bsrc[(size_t)(koff + kk) * NOUT + n];
        }

        __syncthreads();

        // ---- A fragment (contiguous k order: k = half*16 + s, matching B) ----
        v16bf afrag;
        const __bf16* arow = &Alds[(wv * 16 + m16) * 36];
#pragma unroll
        for (int v = 0; v < 8; ++v) {
            int kk = half * 16 + 2 * v;
            afrag[2 * v]     = arow[kk];
            afrag[2 * v + 1] = arow[kk + 1];
        }

        // ---- B fragments in batches of 4, then back-to-back WMMAs ----
#pragma unroll
        for (int nt0 = 0; nt0 < NT; nt0 += 4) {
            v16bf bfr[4];
#pragma unroll
            for (int j = 0; j < 4; ++j) {
                const __bf16* bcol = &Blds[((nt0 + j) * 16 + m16) * 34 + half * 16];
#pragma unroll
                for (int v = 0; v < 16; ++v) bfr[j][v] = bcol[v];
            }
#pragma unroll
            for (int j = 0; j < 4; ++j)
                acc[nt0 + j] = __builtin_amdgcn_wmma_f32_16x16x32_bf16(
                    false, afrag, false, bfr[j], (short)0, acc[nt0 + j], false, false);
        }

        __syncthreads();
    }

    // ---- epilogue (C/D f32 16x16 layout: VGPR r -> row half*8+r, col m16) ----
    if (EPI == 0) {                       // bias + ReLU
#pragma unroll
        for (int nt = 0; nt < NT; ++nt) {
            float b = bias[nt * 16 + m16];
#pragma unroll
            for (int r = 0; r < 8; ++r) {
                int grow = mbase + wv * 16 + half * 8 + r;
                float val = fmaxf(acc[nt][r] + b, 0.0f);
                if (grow < nrows)
                    C[(size_t)grow * NOUT + nt * 16 + m16] = val;
            }
        }
    } else if (EPI == 1) {                // plain store
#pragma unroll
        for (int nt = 0; nt < NT; ++nt) {
#pragma unroll
            for (int r = 0; r < 8; ++r) {
                int grow = mbase + wv * 16 + half * 8 + r;
                if (grow < nrows)
                    C[(size_t)grow * NOUT + nt * 16 + m16] = acc[nt][r];
            }
        }
    } else {                              // bias + addend + row softmax (NOUT=64)
#pragma unroll
        for (int r = 0; r < 8; ++r) {
            int grow = mbase + wv * 16 + half * 8 + r;
            bool valid = grow < nrows;
            float v[NT];
#pragma unroll
            for (int nt = 0; nt < NT; ++nt) {
                int col = nt * 16 + m16;
                float a = valid ? Cadd[(size_t)grow * 64 + col] : 0.0f;
                v[nt] = acc[nt][r] + a + bias[col];
            }
            float m = v[0];
#pragma unroll
            for (int nt = 1; nt < NT; ++nt) m = fmaxf(m, v[nt]);
#pragma unroll
            for (int off = 8; off > 0; off >>= 1)       // within 16-lane half = one row
                m = fmaxf(m, __shfl_xor(m, off, 32));
            float e[NT], s = 0.0f;
#pragma unroll
            for (int nt = 0; nt < NT; ++nt) { e[nt] = __expf(v[nt] - m); s += e[nt]; }
#pragma unroll
            for (int off = 8; off > 0; off >>= 1)
                s += __shfl_xor(s, off, 32);
            float inv = 1.0f / s;
            if (valid) {
#pragma unroll
                for (int nt = 0; nt < NT; ++nt)
                    C[(size_t)grow * 64 + nt * 16 + m16] = e[nt] * inv;
            }
        }
    }
}

// ---------------------------------------------------------------------------
extern "C" void kernel_launch(void* const* d_in, const int* in_sizes, int n_in,
                              void* d_out, int out_size, void* d_ws, size_t ws_size,
                              hipStream_t stream)
{
    (void)in_sizes; (void)n_in; (void)out_size; (void)ws_size;

    const float*     x     = (const float*)d_in[0];
    const long long* edges = (const long long*)d_in[1];   // int64 [2][E]
    const float*     W_l1  = (const float*)d_in[2];
    const float*     b_l1  = (const float*)d_in[3];
    const float*     W_r1  = (const float*)d_in[4];
    const float*     W_l2  = (const float*)d_in[5];
    const float*     b_l2  = (const float*)d_in[6];
    const float*     W_r2  = (const float*)d_in[7];
    float* out = (float*)d_out;

    const long long* src = edges;
    const long long* dst = edges + N_EDGES;

    // ws: region0 [N][128] = agg128, later reused as y2 [N][64] + agg64 [N][64]
    //     region1 [N][128] = h
    float* agg128 = (float*)d_ws;
    float* h      = agg128 + (size_t)N_NODES * HID;
    float* y2     = agg128;                               // [N][64]
    float* agg64  = agg128 + (size_t)N_NODES * OUT_DIM;   // [N][64]

    const int scatter_blocks = (N_EDGES * 32 + 255) / 256;
    const int gemm_blocks    = (N_NODES + 63) / 64;

    // Layer 1: agg1 = sum x[src];  h = relu([agg1|x]@[Wl1;Wr1] + b1)
    hipMemsetAsync(agg128, 0, sizeof(float) * (size_t)N_NODES * HID, stream);
    sage_scatter_add<128><<<scatter_blocks, 256, 0, stream>>>(x, src, dst, agg128, N_EDGES);
    sage_gemm<128, 2, 0><<<gemm_blocks, 128, 0, stream>>>(
        agg128, x, W_l1, W_r1, b_l1, nullptr, h, N_NODES);

    // Layer 2 (projection commuted before aggregation to halve atomics):
    //   y2 = h @ Wl2;  agg64 = sum y2[src];  out = softmax(h @ Wr2 + agg64 + b2)
    sage_gemm<64, 1, 1><<<gemm_blocks, 128, 0, stream>>>(
        h, nullptr, W_l2, nullptr, nullptr, nullptr, y2, N_NODES);
    hipMemsetAsync(agg64, 0, sizeof(float) * (size_t)N_NODES * OUT_DIM, stream);
    sage_scatter_add<64><<<scatter_blocks, 256, 0, stream>>>(y2, src, dst, agg64, N_EDGES);
    sage_gemm<64, 1, 2><<<gemm_blocks, 128, 0, stream>>>(
        h, nullptr, W_r2, nullptr, b_l2, agg64, out, N_NODES);
}
